// MaxFlowGIN_50087908605956
// MI455X (gfx1250) — matrix operand order repaired and checked by reading, not compiled, not run on hardware
//
#include <hip/hip_runtime.h>
#include <hip/hip_bf16.h>

// ---------------------------------------------------------------------------
// Types for CDNA5 WMMA (wave32)
// ---------------------------------------------------------------------------
typedef __attribute__((ext_vector_type(8)))  _Float16 v8h;
typedef __attribute__((ext_vector_type(16))) _Float16 v16h;
typedef __attribute__((ext_vector_type(8)))  float    v8f;

#define NEG_SLOPE 0.01f

__device__ inline v8f wmma_f16(v16h a, v16h b, v8f c) {
    // D = A(16x32 f16) * B(32x16 f16) + C(16x16 f32)
    return __builtin_amdgcn_wmma_f32_16x16x32_f16(
        /*neg_a=*/false, a, /*neg_b=*/false, b,
        /*c_mod=*/(short)0, c, /*reuse_a=*/false, /*reuse_b=*/false);
}

__device__ inline v16h make_v16(v8h lo, v8h hi) {
    v16h a;
#pragma unroll
    for (int i = 0; i < 8; ++i) { a[i] = lo[i]; a[i + 8] = hi[i]; }
    return a;
}

// A-operand (16x32 f16) load from a row-major [16 x ldk] f16 tile in LDS.
// ISA layout: lane<16 -> row=lane,   K = kbase+0..7  and kbase+16..23
//             lane>=16 -> row=lane-16, K = kbase+8..15 and kbase+24..31
__device__ inline v16h load_a_lds(const _Float16* base, int lane, int kbase, int ldk) {
    int row = lane & 15;
    int kb  = kbase + ((lane >> 4) << 3);
    const _Float16* r = base + row * ldk + kb;
    v8h lo = *(const v8h*)(r);
    v8h hi = *(const v8h*)(r + 16);
    return make_v16(lo, hi);
}

// B-operand (32x16 f16) from pre-packed weights: 16 contiguous halfs per lane,
// tile (kt,nt) occupies 32*16 halfs.
__device__ inline v16h load_b_packed(const _Float16* pw, int lane, int kt, int nt, int ntiles) {
    const _Float16* q = pw + ((((size_t)kt * ntiles + nt) * 32 + lane) << 4);
    const v8h* p = (const v8h*)q;
    return make_v16(p[0], p[1]);
}

// ---------------------------------------------------------------------------
// Second MLP GEMM: hout[16 x DOUT] = relu( A_lds[16 x DMID] @ Wb + bb )
// ---------------------------------------------------------------------------
template<int DMID, int DOUT>
__device__ inline void mlp2_store(const _Float16* sMw, int lane,
                                  const _Float16* __restrict__ pWb,
                                  const float* __restrict__ bb,
                                  float* __restrict__ hout,
                                  int rowbase, int N) {
    constexpr int NT = DOUT / 16;
    constexpr int KS = DMID / 32;
    v8f zero = {};
    v8f acc[NT];
#pragma unroll
    for (int nt = 0; nt < NT; ++nt) acc[nt] = zero;
#pragma unroll
    for (int ks = 0; ks < KS; ++ks) {
        v16h a = load_a_lds(sMw, lane, ks * 32, DMID);
#pragma unroll
        for (int nt = 0; nt < NT; ++nt) {
            v16h b = load_b_packed(pWb, lane, ks, nt, NT);
            acc[nt] = wmma_f16(a, b, acc[nt]);
        }
    }
    // C/D layout: lane<16: col=lane, rows v+0..7 ; lane>=16: col=lane-16, rows v+8..15
    int col0 = lane & 15;
    int rofs = (lane >> 4) << 3;
    if (rowbase + 16 <= N) {
        // Fast path (wave-uniform): full tile, branch-free stores
#pragma unroll
        for (int nt = 0; nt < NT; ++nt) {
            int col = nt * 16 + col0;
            float bias = bb[col];
#pragma unroll
            for (int v = 0; v < 8; ++v) {
                float val = acc[nt][v] + bias;
                val = val > 0.f ? val : 0.f;  // ReLU
                hout[(size_t)(rowbase + rofs + v) * DOUT + col] = val;
            }
        }
    } else {
#pragma unroll
        for (int nt = 0; nt < NT; ++nt) {
            int col = nt * 16 + col0;
            float bias = bb[col];
#pragma unroll
            for (int v = 0; v < 8; ++v) {
                int row = rowbase + rofs + v;
                if (row < N) {
                    float val = acc[nt][v] + bias;
                    val = val > 0.f ? val : 0.f;
                    hout[(size_t)row * DOUT + col] = val;
                }
            }
        }
    }
}

// ---------------------------------------------------------------------------
// Fused GIN layer (layers 2,3): h_out = relu( leaky(( h+agg ) @ Wa + ba) @ Wb + bb )
// Block = 128 threads = 4 waves; each wave owns a 16-row node tile.
// ---------------------------------------------------------------------------
template<int DIN, int DMID, int DOUT>
__global__ void __launch_bounds__(128)
gin_layer_kernel(const float* __restrict__ hin, const float* __restrict__ agg,
                 const _Float16* __restrict__ pWa, const float* __restrict__ ba,
                 const _Float16* __restrict__ pWb, const float* __restrict__ bb,
                 float* __restrict__ hout, int N) {
    __shared__ _Float16 sA[4][16 * DIN];
    __shared__ _Float16 sM[4][16 * DMID];
    int wid = threadIdx.x >> 5;
    int lane = threadIdx.x & 31;
    int rowbase = blockIdx.x * 64 + wid * 16;

    // Stage 0: A = fp16(h + agg), row-major in LDS (coalesced across lanes)
    if (rowbase + 16 <= N) {
#pragma unroll 4
        for (int idx = lane; idx < 16 * DIN; idx += 32) {
            int r = idx / DIN, c = idx % DIN;
            size_t g = (size_t)(rowbase + r) * DIN + c;
            sA[wid][idx] = (_Float16)(hin[g] + agg[g]);
        }
    } else {
        for (int idx = lane; idx < 16 * DIN; idx += 32) {
            int r = idx / DIN, c = idx % DIN;
            int gr = rowbase + r;
            float v = 0.f;
            if (gr < N) v = hin[(size_t)gr * DIN + c] + agg[(size_t)gr * DIN + c];
            sA[wid][idx] = (_Float16)v;
        }
    }
    __syncthreads();

    // Stage 1: mid = leaky_relu(A @ Wa + ba) -> LDS fp16
    constexpr int NT1 = DMID / 16;
    constexpr int KS1 = DIN / 32;
    {
        v8f zero = {};
        v8f acc[NT1];
#pragma unroll
        for (int nt = 0; nt < NT1; ++nt) acc[nt] = zero;
#pragma unroll
        for (int ks = 0; ks < KS1; ++ks) {
            v16h a = load_a_lds(sA[wid], lane, ks * 32, DIN);
#pragma unroll
            for (int nt = 0; nt < NT1; ++nt) {
                v16h b = load_b_packed(pWa, lane, ks, nt, NT1);
                acc[nt] = wmma_f16(a, b, acc[nt]);
            }
        }
        int col0 = lane & 15;
        int rofs = (lane >> 4) << 3;
#pragma unroll
        for (int nt = 0; nt < NT1; ++nt) {
            float bias = ba[nt * 16 + col0];
#pragma unroll
            for (int v = 0; v < 8; ++v) {
                float x = acc[nt][v] + bias;
                x = x >= 0.f ? x : NEG_SLOPE * x;  // LeakyReLU
                sM[wid][(rofs + v) * DMID + nt * 16 + col0] = (_Float16)x;
            }
        }
    }
    __syncthreads();

    // Stage 2: h_out = relu(mid @ Wb + bb)
    mlp2_store<DMID, DOUT>(sM[wid], lane, pWb, bb, hout, rowbase, N);
}

// ---------------------------------------------------------------------------
// Layer 1: input dim = 1 (first GEMM is a VALU outer product), second GEMM WMMA
// ---------------------------------------------------------------------------
__global__ void __launch_bounds__(128)
gin_layer1_kernel(const float* __restrict__ x, const float* __restrict__ agg,
                  const float* __restrict__ W1a, const float* __restrict__ b1a,
                  const _Float16* __restrict__ pW1b, const float* __restrict__ b1b,
                  float* __restrict__ h1, int N) {
    __shared__ _Float16 sM[4][16 * 64];
    __shared__ float    sS[4][16];
    int wid = threadIdx.x >> 5;
    int lane = threadIdx.x & 31;
    int rowbase = blockIdx.x * 64 + wid * 16;

    if (lane < 16) {
        int gr = rowbase + lane;
        sS[wid][lane] = (gr < N) ? (x[gr] + agg[gr]) : 0.f;
    }
    __syncthreads();

    // mid[r][c] = leaky(s[r]*W1a[c] + b1a[c]),  16x64
#pragma unroll
    for (int i = 0; i < 32; ++i) {
        int idx = lane + 32 * i;
        int r = idx >> 6, c = idx & 63;
        float t = sS[wid][r] * W1a[c] + b1a[c];
        t = t >= 0.f ? t : NEG_SLOPE * t;
        sM[wid][idx] = (_Float16)t;
    }
    __syncthreads();

    mlp2_store<64, 64>(sM[wid], lane, pW1b, b1b, h1, rowbase, N);
}

// ---------------------------------------------------------------------------
// Edge scatter-add: agg[dst] += h[src]  (the bandwidth/atomic-bound part)
// ---------------------------------------------------------------------------
template<int D>
__global__ void scatter_kernel(const float* __restrict__ h,
                               const int* __restrict__ src,
                               const int* __restrict__ dst,
                               float* __restrict__ agg, int E) {
    if constexpr (D == 1) {
        int t = blockIdx.x * blockDim.x + threadIdx.x;
        if (t < E) atomicAdd(&agg[dst[t]], h[src[t]]);
    } else {
        constexpr int C = D / 4;  // float4 chunks per edge
        int t = blockIdx.x * blockDim.x + threadIdx.x;
        if (t < E * C) {
            int e = t / C, c4 = t % C;
            int s = src[e], d = dst[e];
            float4 v = ((const float4*)(h + (size_t)s * D))[c4];
            float* a = agg + (size_t)d * D + c4 * 4;
            atomicAdd(a + 0, v.x);
            atomicAdd(a + 1, v.y);
            atomicAdd(a + 2, v.z);
            atomicAdd(a + 3, v.w);
        }
    }
}

// ---------------------------------------------------------------------------
// Weight pre-pack into WMMA B-operand order (f32 [fin x fout] -> f16 packed).
// Packed element t: h=t&15, l=(t>>4)&31, tile=t>>9, nt=tile%NT, kt=tile/NT
//   K = kt*32 + (h&7) + (h>=8 ? 16 : 0) + (l>=16 ? 8 : 0),  Ncol = nt*16 + (l&15)
// ---------------------------------------------------------------------------
__global__ void pack_weight_kernel(const float* __restrict__ W, _Float16* __restrict__ out,
                                   int fin, int fout) {
    int t = blockIdx.x * blockDim.x + threadIdx.x;
    if (t < fin * fout) {
        int h = t & 15;
        int l = (t >> 4) & 31;
        int tile = t >> 9;
        int ntiles = fout >> 4;
        int nt = tile % ntiles;
        int kt = tile / ntiles;
        int k = kt * 32 + (h & 7) + ((h & 8) << 1) + ((l >> 4) << 3);
        int n = nt * 16 + (l & 15);
        out[t] = (_Float16)W[(size_t)k * fout + n];
    }
}

// ---------------------------------------------------------------------------
// Pool / counts / final FC
// ---------------------------------------------------------------------------
__global__ void zero_kernel(float* __restrict__ p, int n) {
    int t = blockIdx.x * blockDim.x + threadIdx.x;
    if (t < n) p[t] = 0.f;
}

__global__ void pool_kernel(const float* __restrict__ h3, const int* __restrict__ batch,
                            float* __restrict__ pooled, int N) {
    int t = blockIdx.x * blockDim.x + threadIdx.x;
    if (t < N * 256) {
        int node = t >> 8;
        int g = batch[node];
        atomicAdd(&pooled[g * 256 + (t & 255)], h3[t]);
    }
}

__global__ void count_kernel(const int* __restrict__ batch, float* __restrict__ counts, int N) {
    int t = blockIdx.x * blockDim.x + threadIdx.x;
    if (t < N) atomicAdd(&counts[batch[t]], 1.f);
}

__global__ void final_kernel(const float* __restrict__ pooled, const float* __restrict__ counts,
                             const float* __restrict__ Wfc, const float* __restrict__ bfc,
                             float* __restrict__ out) {
    int g = threadIdx.x;
    if (g < 64) {
        float cnt = counts[g] > 1.f ? counts[g] : 1.f;
        float s = 0.f;
        for (int c = 0; c < 256; ++c) s += pooled[g * 256 + c] * Wfc[c];
        out[g] = s / cnt + bfc[0];
    }
}

// ---------------------------------------------------------------------------
// Host launcher
// ---------------------------------------------------------------------------
extern "C" void kernel_launch(void* const* d_in, const int* in_sizes, int n_in,
                              void* d_out, int out_size, void* d_ws, size_t ws_size,
                              hipStream_t stream) {
    const int N = in_sizes[0];
    const int E = in_sizes[1] / 2;

    const float* x     = (const float*)d_in[0];
    const int*   src   = (const int*)d_in[1];
    const int*   dst   = src + E;
    const int*   batch = (const int*)d_in[2];
    const float* W1a = (const float*)d_in[3],  *b1a = (const float*)d_in[4];
    const float* W1b = (const float*)d_in[5],  *b1b = (const float*)d_in[6];
    const float* W2a = (const float*)d_in[7],  *b2a = (const float*)d_in[8];
    const float* W2b = (const float*)d_in[9],  *b2b = (const float*)d_in[10];
    const float* W3a = (const float*)d_in[11], *b3a = (const float*)d_in[12];
    const float* W3b = (const float*)d_in[13], *b3b = (const float*)d_in[14];
    const float* Wfc = (const float*)d_in[15], *bfc = (const float*)d_in[16];
    float* out = (float*)d_out;

    // Workspace carve-up
    char* ws = (char*)d_ws;
    size_t off = 0;
    auto carve = [&](size_t bytes) -> void* {
        void* p = ws + off;
        off = (off + bytes + 255) & ~(size_t)255;
        return p;
    };
    float*     agg    = (float*)carve((size_t)N * 128 * 4);
    float*     h1     = (float*)carve((size_t)N * 64 * 4);
    float*     h2     = (float*)carve((size_t)N * 128 * 4);
    float*     h3     = (float*)carve((size_t)N * 256 * 4);
    float*     pooled = (float*)carve(64 * 256 * 4);
    float*     counts = (float*)carve(64 * 4);
    _Float16*  pW1b   = (_Float16*)carve(64 * 64 * 2);
    _Float16*  pW2a   = (_Float16*)carve(64 * 128 * 2);
    _Float16*  pW2b   = (_Float16*)carve(128 * 128 * 2);
    _Float16*  pW3a   = (_Float16*)carve(128 * 256 * 2);
    _Float16*  pW3b   = (_Float16*)carve(256 * 256 * 2);
    (void)ws_size; (void)n_in; (void)out_size;

    const int TB = 256;
    auto blocks = [&](long long n) { return (unsigned)((n + TB - 1) / TB); };

    // Pre-pack weights into WMMA B-operand layout
    pack_weight_kernel<<<blocks(64 * 64),   TB, 0, stream>>>(W1b, pW1b, 64, 64);
    pack_weight_kernel<<<blocks(64 * 128),  TB, 0, stream>>>(W2a, pW2a, 64, 128);
    pack_weight_kernel<<<blocks(128 * 128), TB, 0, stream>>>(W2b, pW2b, 128, 128);
    pack_weight_kernel<<<blocks(128 * 256), TB, 0, stream>>>(W3a, pW3a, 128, 256);
    pack_weight_kernel<<<blocks(256 * 256), TB, 0, stream>>>(W3b, pW3b, 256, 256);

    unsigned gNodes = (unsigned)((N + 63) / 64);  // 4 waves x 16 rows per block

    // Layer 1: agg(x) -> MLP(1->64->64) -> relu
    zero_kernel<<<blocks(N), TB, 0, stream>>>(agg, N);
    scatter_kernel<1><<<blocks(E), TB, 0, stream>>>(x, src, dst, agg, E);
    gin_layer1_kernel<<<gNodes, 128, 0, stream>>>(x, agg, W1a, b1a, pW1b, b1b, h1, N);

    // Layer 2: agg(h1) -> MLP(64->128->128) -> relu
    zero_kernel<<<blocks((long long)N * 64), TB, 0, stream>>>(agg, N * 64);
    scatter_kernel<64><<<blocks((long long)E * 16), TB, 0, stream>>>(h1, src, dst, agg, E);
    gin_layer_kernel<64, 128, 128><<<gNodes, 128, 0, stream>>>(h1, agg, pW2a, b2a, pW2b, b2b, h2, N);

    // Layer 3: agg(h2) -> MLP(128->256->256) -> relu
    zero_kernel<<<blocks((long long)N * 128), TB, 0, stream>>>(agg, N * 128);
    scatter_kernel<128><<<blocks((long long)E * 32), TB, 0, stream>>>(h2, src, dst, agg, E);
    gin_layer_kernel<128, 256, 256><<<gNodes, 128, 0, stream>>>(h2, agg, pW3a, b3a, pW3b, b3b, h3, N);

    // Global mean pool + FC
    zero_kernel<<<blocks(64 * 256), TB, 0, stream>>>(pooled, 64 * 256);
    zero_kernel<<<1, 64, 0, stream>>>(counts, 64);
    pool_kernel<<<blocks((long long)N * 256), TB, 0, stream>>>(h3, batch, pooled, N);
    count_kernel<<<blocks(N), TB, 0, stream>>>(batch, counts, N);
    final_kernel<<<1, 64, 0, stream>>>(pooled, counts, Wfc, bfc, out);
}
